// LSHSelfAttention_30906584662594
// MI455X (gfx1250) — matrix operand order repaired and checked by reading, not compiled, not run on hardware
//
#include <hip/hip_runtime.h>
#include <math.h>

#define D_MODEL 1024
#define NHEAD   16
#define DK      64
#define SEQ     2048
#define LDT     40   // attention P-tile LDS row stride (bf16): 80B, 16B-aligned
#define LDG     80   // GEMM LDS row stride (bf16 elems): 160B, 16B-aligned, 40-bank step

typedef __bf16 bf16_t;
typedef bf16_t v2bf  __attribute__((ext_vector_type(2)));
typedef bf16_t v8bf  __attribute__((ext_vector_type(8)));
typedef bf16_t v16bf __attribute__((ext_vector_type(16)));
typedef float  v8f   __attribute__((ext_vector_type(8)));
typedef float  v4f   __attribute__((ext_vector_type(4)));

static __device__ inline v16bf make_frag(v8bf lo, v8bf hi) {
  v16bf f;
#pragma unroll
  for (int i = 0; i < 8; ++i) { f[i] = lo[i]; f[i + 8] = hi[i]; }
  return f;
}

// A-operand fragment (16x32 bf16). `row` = this lane's 32 contiguous k-values.
// Lane half g holds k-runs [8g,8g+8) and [16+8g,24+8g) (CDNA5 16-bit A layout).
static __device__ inline v16bf load_afrag(const bf16_t* row, int g) {
  v8bf lo = *(const v8bf*)(row + 8 * g);
  v8bf hi = *(const v8bf*)(row + 16 + 8 * g);
  return make_frag(lo, hi);
}

// B-operand fragment (32x16 bf16). `col` = this lane's column, 32 contiguous
// k-values. Lane half g holds k-runs [8g,8g+8) and [16+8g,24+8g) (same as A).
static __device__ inline v16bf load_bfrag(const bf16_t* col, int g) {
  v8bf lo = *(const v8bf*)(col + 8 * g);
  v8bf hi = *(const v8bf*)(col + 16 + 8 * g);
  return make_frag(lo, hi);
}

static __device__ inline v8f zero8() {
  v8f z;
#pragma unroll
  for (int i = 0; i < 8; ++i) z[i] = 0.f;
  return z;
}

// C = (X @ W^T + bias) * scale, fp32 in, bf16 out, written in attention layout.
// TRANSPOSED==0: Y[(b*H+h)*SEQ + l][d]   (Q, K)
// TRANSPOSED==1: Y[(b*H+h)*DK  + d][l]   (V^T: PV B-operand becomes contiguous-K)
// scale = 1/sqrt(dk) for Q (exact power of two in bf16), 1.0 for K/V.
template <int TRANSPOSED>
__global__ __launch_bounds__(256)
void proj_gemm_bf16(const float* __restrict__ X, const float* __restrict__ W,
                    const float* __restrict__ bias, bf16_t* __restrict__ Y,
                    float scale) {
  __shared__ bf16_t As[128 * LDG];   // 20 KB
  __shared__ bf16_t Bs[128 * LDG];   // 20 KB

  const int t    = threadIdx.x;
  const int lane = t & 31;
  const int wave = t >> 5;
  const int g    = lane >> 4;
  const int ln   = lane & 15;
  const int waveM = (wave >> 1) * 32;  // 4 waves along M
  const int waveN = (wave & 1) * 64;   // 2 waves along N

  const int mBase = blockIdx.y * 128;
  const int nBase = blockIdx.x * 128;

  const int lrow = t >> 1;        // 0..127: tile row this thread stages
  const int lkh  = (t & 1) * 32;  // k-half within 64-wide K stage

  v8f acc[2][4];
#pragma unroll
  for (int i = 0; i < 2; ++i)
#pragma unroll
    for (int j = 0; j < 4; ++j) acc[i][j] = zero8();

  const float* xrow = X + (size_t)(mBase + lrow) * D_MODEL + lkh;
  const float* wrow = W + (size_t)(nBase + lrow) * D_MODEL + lkh;

  for (int k0 = 0; k0 < D_MODEL; k0 += 64) {
    // cooperative fp32 load -> bf16 convert -> LDS stage (64-wide K stage)
    const v4f* pa = (const v4f*)(xrow + k0);
    const v4f* pb = (const v4f*)(wrow + k0);
    v4f a[8], w4[8];
#pragma unroll
    for (int i = 0; i < 8; ++i) { a[i] = pa[i]; w4[i] = pb[i]; }
    v8bf ha[4], hb[4];
#pragma unroll
    for (int q = 0; q < 4; ++q)
#pragma unroll
      for (int i = 0; i < 8; ++i) {
        ha[q][i] = (bf16_t)a[2 * q + (i >> 2)][i & 3];
        hb[q][i] = (bf16_t)w4[2 * q + (i >> 2)][i & 3];
      }
    __syncthreads();  // previous stage's fragment reads done
#pragma unroll
    for (int q = 0; q < 4; ++q) {
      *(v8bf*)&As[lrow * LDG + lkh + 8 * q] = ha[q];
      *(v8bf*)&Bs[lrow * LDG + lkh + 8 * q] = hb[q];
    }
    __syncthreads();

#pragma unroll
    for (int s2 = 0; s2 < 64; s2 += 32) {
      v16bf af[2], bfv[4];
#pragma unroll
      for (int i = 0; i < 2; ++i)
        af[i] = load_afrag(&As[(waveM + i * 16 + ln) * LDG + s2], g);
#pragma unroll
      for (int j = 0; j < 4; ++j)
        bfv[j] = load_bfrag(&Bs[(waveN + j * 16 + ln) * LDG + s2], g);
#pragma unroll
      for (int i = 0; i < 2; ++i)
#pragma unroll
        for (int j = 0; j < 4; ++j)
          acc[i][j] = __builtin_amdgcn_wmma_f32_16x16x32_bf16(
              false, af[i], false, bfv[j], (short)0, acc[i][j], false, false);
    }
  }

  // Epilogue. A 128-row M block never crosses a batch boundary (2048 % 128 == 0),
  // so the batch index is block-uniform; one base pointer per (i,j) tile,
  // stores use immediate offsets.
  const int bb    = mBase >> 11;                        // m / SEQ (scalar)
  const int lBase = (mBase & (SEQ - 1)) + waveM + 8 * g;
#pragma unroll
  for (int j = 0; j < 4; ++j) {
    const int n = nBase + waveN + j * 16 + ln;
    const float bv = bias[n];
    const int h = n >> 6, d = n & (DK - 1);
    bf16_t* yb;
    if (TRANSPOSED)
      yb = Y + (((size_t)(bb * NHEAD + h)) * DK + d) * SEQ + lBase;
    else
      yb = Y + (((size_t)(bb * NHEAD + h)) * SEQ + lBase) * DK + d;
#pragma unroll
    for (int i = 0; i < 2; ++i) {
      if (TRANSPOSED) {
        // adjacent v -> adjacent l: pack bf16 pairs, b32 stores (4B aligned)
#pragma unroll
        for (int v = 0; v < 8; v += 2) {
          v2bf p;
          p[0] = (bf16_t)((acc[i][j][v] + bv) * scale);
          p[1] = (bf16_t)((acc[i][j][v + 1] + bv) * scale);
          *(v2bf*)(yb + i * 16 + v) = p;
        }
      } else {
#pragma unroll
        for (int v = 0; v < 8; ++v)
          yb[(i * 16 + v) * DK] = (bf16_t)((acc[i][j][v] + bv) * scale);
      }
    }
  }
}

// Flash-style streaming LSH attention: 1 block = (b,h, 128 q-rows); 4 waves,
// each wave owns a 32-row tile (2 WMMA M-tiles) so every K / V^T B-fragment
// is reused by two WMMAs: 16 wmma per 32-key block from 16 b128 loads.
__global__ __launch_bounds__(128)
void lsh_attention(const bf16_t* __restrict__ Qb, const bf16_t* __restrict__ Kb,
                   const bf16_t* __restrict__ VTb, const int* __restrict__ hashq,
                   const int* __restrict__ hashk, float* __restrict__ out) {
  __shared__ bf16_t Pl[4][32 * LDT];  // per-wave 32x32 P tile

  const int bh   = blockIdx.y;
  const int b    = bh >> 4;
  const int h    = bh & (NHEAD - 1);
  const int wave = threadIdx.x >> 5;
  const int lane = threadIdx.x & 31;
  const int g    = lane >> 4;
  const int ln   = lane & 15;
  const int qBase = blockIdx.x * 128 + wave * 32;

  const bf16_t* Q  = Qb  + (size_t)bh * SEQ * DK;
  const bf16_t* K  = Kb  + (size_t)bh * SEQ * DK;
  const bf16_t* VT = VTb + (size_t)bh * DK * SEQ;
  const int* hqp = hashq + (size_t)bh * SEQ + qBase;
  const int* hkp = hashk + (size_t)bh * SEQ;

  // Q fragments (pre-scaled by 1/sqrt(dk) in the projection) + row hashes
  v16bf qf[2][2];
  int hqv[2][8];
#pragma unroll
  for (int mt = 0; mt < 2; ++mt) {
#pragma unroll
    for (int s = 0; s < 2; ++s)
      qf[mt][s] = load_afrag(Q + (size_t)(qBase + mt * 16 + ln) * DK + s * 32, g);
#pragma unroll
    for (int v = 0; v < 8; ++v) hqv[mt][v] = hqp[mt * 16 + 8 * g + v];
  }

  v8f o[2][4];
  float mrun[2][8], srun[2][8];
#pragma unroll
  for (int mt = 0; mt < 2; ++mt) {
#pragma unroll
    for (int j = 0; j < 4; ++j) o[mt][j] = zero8();
#pragma unroll
    for (int v = 0; v < 8; ++v) { mrun[mt][v] = -3.0e38f; srun[mt][v] = 0.f; }
  }

  bf16_t* P = &Pl[wave][0];

  for (int kb = 0; kb < SEQ; kb += 32) {
    // pull next key block toward the WGP while this block's WMMAs run
    // (unguarded: past the K region it speculatively touches the adjacent
    //  workspace tensor -- valid memory, harmless)
    __builtin_prefetch(K + (size_t)(kb + 32 + lane) * DK, 0, 3);

    // S = Q K^T : 2 M-tiles x 2 n-tiles, K fragments reused across M-tiles
    v8f sc[2][2];
    {
      v16bf kf[2][2];
#pragma unroll
      for (int nt = 0; nt < 2; ++nt) {
        const bf16_t* krow = K + (size_t)(kb + nt * 16 + ln) * DK;
#pragma unroll
        for (int s = 0; s < 2; ++s) kf[nt][s] = load_bfrag(krow + s * 32, g);
      }
#pragma unroll
      for (int mt = 0; mt < 2; ++mt)
#pragma unroll
        for (int nt = 0; nt < 2; ++nt) {
          sc[mt][nt] = zero8();
#pragma unroll
          for (int s = 0; s < 2; ++s)
            sc[mt][nt] = __builtin_amdgcn_wmma_f32_16x16x32_bf16(
                false, qf[mt][s], false, kf[nt][s], (short)0, sc[mt][nt],
                false, false);
        }
    }

    int hkv[2];
#pragma unroll
    for (int nt = 0; nt < 2; ++nt) hkv[nt] = hkp[kb + nt * 16 + ln];

    // per M-tile: LSH mask + online softmax (width-16 xor-shuffle reductions)
#pragma unroll
    for (int mt = 0; mt < 2; ++mt) {
#pragma unroll
      for (int nt = 0; nt < 2; ++nt)
#pragma unroll
        for (int v = 0; v < 8; ++v) {
          float x = sc[mt][nt][v];
          if (hqv[mt][v] != hkv[nt]) x -= 1.0e9f;
          sc[mt][nt][v] = x;
        }
      float tmax[8], alpha[8], psum[8];
#pragma unroll
      for (int v = 0; v < 8; ++v) tmax[v] = fmaxf(sc[mt][0][v], sc[mt][1][v]);
#pragma unroll
      for (int v = 0; v < 8; ++v)
#pragma unroll
        for (int msk = 1; msk < 16; msk <<= 1)
          tmax[v] = fmaxf(tmax[v], __shfl_xor(tmax[v], msk, 16));
#pragma unroll
      for (int v = 0; v < 8; ++v) {
        const float mnew = fmaxf(mrun[mt][v], tmax[v]);
        alpha[v] = __expf(mrun[mt][v] - mnew);
        mrun[mt][v] = mnew;
        psum[v] = 0.f;
      }
#pragma unroll
      for (int nt = 0; nt < 2; ++nt)
#pragma unroll
        for (int v = 0; v < 8; ++v) {
          const float p = __expf(sc[mt][nt][v] - mrun[mt][v]);
          sc[mt][nt][v] = p;
          psum[v] += p;
        }
#pragma unroll
      for (int v = 0; v < 8; ++v)
#pragma unroll
        for (int msk = 1; msk < 16; msk <<= 1)
          psum[v] += __shfl_xor(psum[v], msk, 16);
#pragma unroll
      for (int v = 0; v < 8; ++v)
        srun[mt][v] = srun[mt][v] * alpha[v] + psum[v];
#pragma unroll
      for (int j = 0; j < 4; ++j)
#pragma unroll
        for (int v = 0; v < 8; ++v) o[mt][j][v] *= alpha[v];

      // P: C-layout -> A-layout via per-wave LDS round-trip
#pragma unroll
      for (int nt = 0; nt < 2; ++nt)
#pragma unroll
        for (int v = 0; v < 8; ++v)
          P[(mt * 16 + 8 * g + v) * LDT + nt * 16 + ln] = (bf16_t)sc[mt][nt][v];
    }
    asm volatile("s_wait_dscnt 0" ::: "memory");  // wave-local LDS RAW fence
    v16bf pf[2];
#pragma unroll
    for (int mt = 0; mt < 2; ++mt)
      pf[mt] = load_afrag(P + (mt * 16 + ln) * LDT, g);

    // O += P @ V : V^T fragments reused across both M-tiles
#pragma unroll
    for (int j = 0; j < 4; ++j) {
      const bf16_t* vcol = VT + (size_t)(j * 16 + ln) * SEQ + kb;
      __builtin_prefetch(vcol + 32, 0, 3);
      const v16bf vf = load_bfrag(vcol, g);
#pragma unroll
      for (int mt = 0; mt < 2; ++mt)
        o[mt][j] = __builtin_amdgcn_wmma_f32_16x16x32_bf16(
            false, pf[mt], false, vf, (short)0, o[mt][j], false, false);
    }
  }

  // normalize + merge heads: out[b][l][h*64+d], fp32, immediate-offset stores
#pragma unroll
  for (int mt = 0; mt < 2; ++mt) {
    float rinv[8];
#pragma unroll
    for (int v = 0; v < 8; ++v) rinv[v] = __frcp_rn(srun[mt][v]);
    float* ob = out + ((size_t)b * SEQ + qBase + mt * 16 + 8 * g) * D_MODEL +
                h * DK + ln;
#pragma unroll
    for (int j = 0; j < 4; ++j)
#pragma unroll
      for (int v = 0; v < 8; ++v)
        ob[v * D_MODEL + j * 16] = o[mt][j][v] * rinv[v];
  }
}

extern "C" void kernel_launch(void* const* d_in, const int* in_sizes, int n_in,
                              void* d_out, int out_size, void* d_ws, size_t ws_size,
                              hipStream_t stream) {
  const float* query = (const float*)d_in[0];
  const float* key   = (const float*)d_in[1];
  const float* value = (const float*)d_in[2];
  const int*   hashq = (const int*)d_in[3];
  const int*   hashk = (const int*)d_in[4];
  const float* Wq = (const float*)d_in[5];
  const float* bq = (const float*)d_in[6];
  const float* Wk = (const float*)d_in[7];
  const float* bk = (const float*)d_in[8];
  const float* Wv = (const float*)d_in[9];
  const float* bv = (const float*)d_in[10];
  float* out = (float*)d_out;

  const int B = in_sizes[0] / (SEQ * D_MODEL);
  const size_t per = (size_t)B * NHEAD * SEQ * DK;  // bf16 elements per tensor
  bf16_t* Qb  = (bf16_t*)d_ws;   // [B*H][SEQ][64], pre-scaled by 1/sqrt(dk)
  bf16_t* Kb  = Qb + per;        // [B*H][SEQ][64]
  bf16_t* VTb = Kb + per;        // [B*H][64][SEQ]

  dim3 gproj(D_MODEL / 128, (B * SEQ) / 128);
  proj_gemm_bf16<0><<<gproj, 256, 0, stream>>>(query, Wq, bq, Qb, 0.125f);
  proj_gemm_bf16<0><<<gproj, 256, 0, stream>>>(key,   Wk, bk, Kb, 1.0f);
  proj_gemm_bf16<1><<<gproj, 256, 0, stream>>>(value, Wv, bv, VTb, 1.0f);

  dim3 gattn(SEQ / 128, B * NHEAD);
  lsh_attention<<<gattn, 128, 0, stream>>>(Qb, Kb, VTb, hashq, hashk, out);
}
